// MS_SSA_936302870975
// MI455X (gfx1250) — matrix operand order repaired and checked by reading, not compiled, not run on hardware
//
#include <hip/hip_runtime.h>
#include <hip/hip_bf16.h>
#include <math.h>

// ---------------------------------------------------------------------------
// MS-SSA forward for MI455X (gfx1250, wave32, WMMA).
// T=512, B=16, N=512, H=8, D=64.  All matmuls via v_wmma_f32_16x16x32_f16.
// Spikes (0/1) are exact in f16; attention GEMMs are bit-exact (f32 accum).
// ---------------------------------------------------------------------------

typedef _Float16 v16h __attribute__((ext_vector_type(16)));
typedef _Float16 h8   __attribute__((ext_vector_type(8)));
typedef float    v8f  __attribute__((ext_vector_type(8)));

#define T_STEPS 512
#define BATCH   16
#define NFEAT   512
#define HEADS   8
#define DHEAD   64
#define CH      8192            // B*N = channels for LIF scans
#define TBN     4194304         // T*B*N
#define NN      262144          // N*N weight elements
#define EPSF    1e-5f

__device__ inline v8f zero8() {
  v8f z;
#pragma unroll
  for (int i = 0; i < 8; ++i) z[i] = 0.0f;
  return z;
}

__device__ inline v16h mk16(h8 lo, h8 hi) {
  v16h r;
#pragma unroll
  for (int i = 0; i < 8; ++i) { r[i] = lo[i]; r[i + 8] = hi[i]; }
  return r;
}

// D = A(16x32 f16) * B(32x16 f16) + C(16x16 f32)
__device__ inline v8f wmma16(v16h a, v16h b, v8f c) {
  return __builtin_amdgcn_wmma_f32_16x16x32_f16(
      /*neg_a=*/false, a, /*neg_b=*/false, b,
      /*c_mod=*/(short)0, c, /*reuse_a=*/false, /*reuse_b=*/false);
}

__device__ inline void lds_fence() {
  asm volatile("s_wait_dscnt 0" ::: "memory");
}

// ---------------------------------------------------------------------------
// Weight conversion f32 -> f16
// ---------------------------------------------------------------------------
__global__ void f32_to_f16_kernel(const float* __restrict__ src,
                                  _Float16* __restrict__ dst, int n) {
  int i = blockIdx.x * blockDim.x + threadIdx.x;
  if (i < n) dst[i] = (_Float16)src[i];
}

// ---------------------------------------------------------------------------
// LayerNorm over feature dim. One block per (t,b) row of 512 features.
// ---------------------------------------------------------------------------
__global__ void ln_kernel(const float* __restrict__ x,
                          const float* __restrict__ g,
                          const float* __restrict__ b,
                          float* __restrict__ xn) {
  __shared__ float s1[256];
  __shared__ float s2[256];
  const int row = blockIdx.x;
  const int tid = threadIdx.x;
  const float* xr = x + (size_t)row * NFEAT;
  float v0 = xr[tid], v1 = xr[tid + 256];
  s1[tid] = v0 + v1;
  s2[tid] = v0 * v0 + v1 * v1;
  __syncthreads();
  for (int off = 128; off > 0; off >>= 1) {
    if (tid < off) { s1[tid] += s1[tid + off]; s2[tid] += s2[tid + off]; }
    __syncthreads();
  }
  const float mean = s1[0] * (1.0f / NFEAT);
  const float var  = s2[0] * (1.0f / NFEAT) - mean * mean;
  const float r    = rsqrtf(var + EPSF);
  float* xo = xn + (size_t)row * NFEAT;
  xo[tid]       = (v0 - mean) * r * g[tid]       + b[tid];
  xo[tid + 256] = (v1 - mean) * r * g[tid + 256] + b[tid + 256];
}

// ---------------------------------------------------------------------------
// Generic multi-step LIF: input viewed as (512 steps, 8192 channels) f32.
// decay=0.5, v_th=1, hard reset to 0.  One lane per channel.
// TROUT=true writes out[c*512 + t] (i.e. (B,H,D,T) layout for V spikes).
// ---------------------------------------------------------------------------
template <typename OutT, bool TROUT>
__global__ void lif_kernel(const float* __restrict__ x, OutT* __restrict__ out) {
  const int c = blockIdx.x * blockDim.x + threadIdx.x;   // 0..8191
  float v = 0.0f;
#pragma unroll 4
  for (int t = 0; t < T_STEPS; ++t) {
    v = v * 0.5f + x[(size_t)t * CH + c];
    const float s = (v - 1.0f) >= 0.0f ? 1.0f : 0.0f;
    const size_t o = TROUT ? ((size_t)c * T_STEPS + t) : ((size_t)t * CH + c);
    out[o] = (OutT)s;
    v *= (1.0f - s);
  }
}

// ---------------------------------------------------------------------------
// Rotary: in = spikes f16 laid out (T,B,N)=(T,B,H,D); out f32 (B,H,T,D).
// ---------------------------------------------------------------------------
__global__ void rotary_kernel(const _Float16* __restrict__ in,
                              float* __restrict__ outp) {
  const int idx = blockIdx.x * blockDim.x + threadIdx.x;   // 0..TBN-1
  const int d = idx & 63;
  const int t = (idx >> 6) & 511;
  const int h = (idx >> 15) & 7;
  const int b = idx >> 18;
  const size_t base = (size_t)t * CH + (size_t)b * NFEAT + h * DHEAD;
  const float x0 = (float)in[base + d];
  const float xp = (float)in[base + (d ^ 1)];
  const int i = d >> 1;
  const float freq = (float)t * powf(10000.0f, -((float)(2 * i) / (float)DHEAD));
  const float cs = cosf(freq), sn = sinf(freq);
  const float rh = (d & 1) ? xp : -xp;
  outp[(size_t)(((b * HEADS + h) * T_STEPS + t)) * DHEAD + d] = x0 * cs + rh * sn;
}

// ---------------------------------------------------------------------------
// GEMM + eval-BN epilogue:  out[r,m] = BN_m( sum_n A[r,n] * W[m,n] )
// A: (8192,512) f16 spikes row-major.  W: (512,512) f16 [m][n].
// bnp: 4x512 f32 rows {gamma,beta,mean,var}.  out: (8192,512) f32.
// One wave computes a 16x64 tile (4 C-frags); 8 waves/block; 512 blocks.
// ---------------------------------------------------------------------------
__global__ void gemm_bn_kernel(const _Float16* __restrict__ A,
                               const _Float16* __restrict__ W,
                               const float* __restrict__ bnp,
                               float* __restrict__ out) {
  const int lane = threadIdx.x & 31;
  const int wv   = threadIdx.x >> 5;
  const int gw   = blockIdx.x * 8 + wv;
  const int rt   = gw >> 3;        // row tile 0..511
  const int cg   = gw & 7;         // 64-col group
  const int half = lane >> 4, l16 = lane & 15;
  const int m0   = cg * 64;
  const _Float16* arow = A + (size_t)(rt * 16 + l16) * NFEAT;

  v8f acc[4];
#pragma unroll
  for (int g = 0; g < 4; ++g) acc[g] = zero8();

  for (int kc = 0; kc < NFEAT; kc += 32) {
    // A-frag: lane row rt*16+l16, chunks [kc+half*8, +8) and [kc+16+half*8, +8)
    const _Float16* ap = arow + kc + half * 8;
    const v16h a = mk16(*(const h8*)ap, *(const h8*)(ap + 16));
#pragma unroll
    for (int g = 0; g < 4; ++g) {
      // B-frag: column m, K chunks [kc+half*8, +8) and [kc+16+half*8, +8) in n of W[m][n]
      const int m = m0 + g * 16 + l16;
      const _Float16* wp = W + (size_t)m * NFEAT + kc + half * 8;
      const v16h bf = mk16(*(const h8*)wp, *(const h8*)(wp + 16));
      acc[g] = wmma16(a, bf, acc[g]);
    }
  }

#pragma unroll
  for (int g = 0; g < 4; ++g) {
    const int m = m0 + g * 16 + l16;
    const float ga = bnp[m], be = bnp[NFEAT + m];
    const float mu = bnp[2 * NFEAT + m], va = bnp[3 * NFEAT + m];
    const float sc = ga * rsqrtf(va + EPSF);
#pragma unroll
    for (int j = 0; j < 8; ++j) {
      const int row = rt * 16 + j + 8 * half;   // C-frag: VGPR j <-> row j+8*half
      out[(size_t)row * NFEAT + m] = (acc[g][j] - mu) * sc + be;
    }
  }
}

// ---------------------------------------------------------------------------
// Fused attention (no softmax, v2 masked):
//   S = Q K^T  (exact counts);  S[:,s]=0 where !mask[b,s];  y = S V * 0.125
// qs/ks: spikes f16 (B,H,T,D).  vtb: V spikes f16 (B,H,D,T).
// ybuf: f32 (T,B,H,D).  Block = (b,h,qtile-of-64), 4 waves x 16 q-rows.
// S(f32 C-frag) -> LDS -> A-frag transpose per wave.
// ---------------------------------------------------------------------------
__global__ void attn_kernel(const _Float16* __restrict__ qs,
                            const _Float16* __restrict__ ks,
                            const _Float16* __restrict__ vtb,
                            const unsigned char* __restrict__ amask,
                            float* __restrict__ ybuf) {
  __shared__ _Float16 stile[4][16][34];   // per-wave 16x32 S tile (+pad)

  const int blk = blockIdx.x;             // 0..1023
  const int qt  = blk & 7;
  const int bh  = blk >> 3;
  const int b   = bh >> 3;
  const int h   = bh & 7;
  const int lane = threadIdx.x & 31;
  const int wv   = threadIdx.x >> 5;
  const int half = lane >> 4, l16 = lane & 15;
  const int t0   = qt * 64 + wv * 16;

  const _Float16* qbh = qs  + (size_t)bh * T_STEPS * DHEAD;
  const _Float16* kbh = ks  + (size_t)bh * T_STEPS * DHEAD;
  const _Float16* vbh = vtb + (size_t)bh * DHEAD * T_STEPS;

  // Q A-frags for this wave's 16 rows, K-dim = d (two 32-wide steps)
  v16h aq[2];
#pragma unroll
  for (int c = 0; c < 2; ++c) {
    const _Float16* qr = qbh + (size_t)(t0 + l16) * DHEAD + c * 32 + half * 8;
    aq[c] = mk16(*(const h8*)qr, *(const h8*)(qr + 16));
  }

  v8f acc[4];
#pragma unroll
  for (int g = 0; g < 4; ++g) acc[g] = zero8();

  for (int s0 = 0; s0 < T_STEPS; s0 += 32) {
    // --- stage 1: S tile (16 q-rows x 32 key-cols), exact integer counts ---
    v8f sacc[2];
    sacc[0] = zero8(); sacc[1] = zero8();
#pragma unroll
    for (int sc = 0; sc < 2; ++sc) {
#pragma unroll
      for (int c = 0; c < 2; ++c) {
        // B-frag: column = key row s0+sc*16+l16, K chunks c*32+half*8 and c*32+16+half*8 in d
        const _Float16* kr =
            kbh + (size_t)(s0 + sc * 16 + l16) * DHEAD + c * 32 + half * 8;
        const v16h bk = mk16(*(const h8*)kr, *(const h8*)(kr + 16));
        sacc[sc] = wmma16(aq[c], bk, sacc[sc]);
      }
    }
    // mask columns, stash S as f16 in LDS (C-frag -> row-major tile)
#pragma unroll
    for (int sc = 0; sc < 2; ++sc) {
      const float mv = amask[b * T_STEPS + s0 + sc * 16 + l16] ? 1.0f : 0.0f;
#pragma unroll
      for (int j = 0; j < 8; ++j)
        stile[wv][j + 8 * half][sc * 16 + l16] = (_Float16)(sacc[sc][j] * mv);
    }
    lds_fence();

    // --- transpose-read S as A-frag (row = l16, K per WMMA A layout) ---
    v16h aS;
#pragma unroll
    for (int e = 0; e < 16; ++e) {
      const int j = e >> 1, p = e & 1;
      const int kk = ((j < 4) ? 0 : 16) + half * 8 + 2 * (j & 3) + p;
      aS[e] = stile[wv][l16][kk];
    }
    lds_fence();

    // --- stage 2: y += S * V ;  V B-frag contiguous from (B,H,D,T) ---
#pragma unroll
    for (int g = 0; g < 4; ++g) {
      const _Float16* vr =
          vbh + (size_t)(g * 16 + l16) * T_STEPS + s0 + half * 8;
      const v16h bv = mk16(*(const h8*)vr, *(const h8*)(vr + 16));
      acc[g] = wmma16(aS, bv, acc[g]);
    }
  }

  // epilogue: scale by 1/sqrt(D)=0.125, write (T,B,H,D)
#pragma unroll
  for (int g = 0; g < 4; ++g) {
    const int d = g * 16 + l16;
#pragma unroll
    for (int j = 0; j < 8; ++j) {
      const int t = t0 + j + 8 * half;
      ybuf[(size_t)t * CH + (size_t)b * NFEAT + h * DHEAD + d] =
          acc[g][j] * 0.125f;
    }
  }
}

// ---------------------------------------------------------------------------
// Host launcher
// ---------------------------------------------------------------------------
extern "C" void kernel_launch(void* const* d_in, const int* in_sizes, int n_in,
                              void* d_out, int out_size, void* d_ws,
                              size_t ws_size, hipStream_t stream) {
  const float* x    = (const float*)d_in[0];
  const unsigned char* amask = (const unsigned char*)d_in[1];
  const float* ln_g = (const float*)d_in[2];
  const float* ln_b = (const float*)d_in[3];
  const float* wq   = (const float*)d_in[4];
  const float* q_bn = (const float*)d_in[5];
  const float* wk   = (const float*)d_in[6];
  const float* k_bn = (const float*)d_in[7];
  const float* wv   = (const float*)d_in[8];
  const float* v_bn = (const float*)d_in[9];
  const float* wp   = (const float*)d_in[10];
  const float* p_bn = (const float*)d_in[11];
  float* out = (float*)d_out;

  char* ws = (char*)d_ws;
  size_t off = 0;
  auto alloc = [&](size_t bytes) -> void* {
    void* p = ws + off;
    off += (bytes + 255) & ~(size_t)255;
    return p;
  };
  _Float16* wq16  = (_Float16*)alloc((size_t)NN * 2);
  _Float16* wk16  = (_Float16*)alloc((size_t)NN * 2);
  _Float16* wv16  = (_Float16*)alloc((size_t)NN * 2);
  _Float16* wp16  = (_Float16*)alloc((size_t)NN * 2);
  float*    xn    = (float*)   alloc((size_t)TBN * 4);   // LN out
  _Float16* xs    = (_Float16*)alloc((size_t)TBN * 2);   // head spikes
  float*    u     = (float*)   alloc((size_t)TBN * 4);   // pre-LIF / y scratch
  float*    rot   = (float*)   alloc((size_t)TBN * 4);   // rotary out (B,H,T,D)
  _Float16* tmp16 = (_Float16*)alloc((size_t)TBN * 2);   // q1/k1/ys spikes
  _Float16* qsb   = (_Float16*)alloc((size_t)TBN * 2);   // Q spikes (B,H,T,D)
  _Float16* ksb   = (_Float16*)alloc((size_t)TBN * 2);   // K spikes (B,H,T,D)
  _Float16* vtb   = (_Float16*)alloc((size_t)TBN * 2);   // V spikes (B,H,D,T)

  // weights -> f16
  f32_to_f16_kernel<<<NN / 256, 256, 0, stream>>>(wq, wq16, NN);
  f32_to_f16_kernel<<<NN / 256, 256, 0, stream>>>(wk, wk16, NN);
  f32_to_f16_kernel<<<NN / 256, 256, 0, stream>>>(wv, wv16, NN);
  f32_to_f16_kernel<<<NN / 256, 256, 0, stream>>>(wp, wp16, NN);

  // LayerNorm + head LIF
  ln_kernel<<<T_STEPS * BATCH, 256, 0, stream>>>(x, ln_g, ln_b, xn);
  lif_kernel<_Float16, false><<<CH / 256, 256, 0, stream>>>(xn, xs);

  // Q branch: Linear+BN -> LIF -> rotary -> LIF
  gemm_bn_kernel<<<512, 256, 0, stream>>>(xs, wq16, q_bn, u);
  lif_kernel<_Float16, false><<<CH / 256, 256, 0, stream>>>(u, tmp16);
  rotary_kernel<<<TBN / 256, 256, 0, stream>>>(tmp16, rot);
  lif_kernel<_Float16, false><<<CH / 256, 256, 0, stream>>>(rot, qsb);

  // K branch
  gemm_bn_kernel<<<512, 256, 0, stream>>>(xs, wk16, k_bn, u);
  lif_kernel<_Float16, false><<<CH / 256, 256, 0, stream>>>(u, tmp16);
  rotary_kernel<<<TBN / 256, 256, 0, stream>>>(tmp16, rot);
  lif_kernel<_Float16, false><<<CH / 256, 256, 0, stream>>>(rot, ksb);

  // V branch (LIF writes transposed (B,H,D,T) for contiguous WMMA B-frags)
  gemm_bn_kernel<<<512, 256, 0, stream>>>(xs, wv16, v_bn, u);
  lif_kernel<_Float16, true><<<CH / 256, 256, 0, stream>>>(u, vtb);

  // Fused masked attention: y = (Q K^T masked) V * scale -> (T,B,H,D) f32
  attn_kernel<<<BATCH * HEADS * 8, 128, 0, stream>>>(qsb, ksb, vtb, amask, u);

  // attn LIF -> proj Linear+BN -> final LIF (f32 out)
  lif_kernel<_Float16, false><<<CH / 256, 256, 0, stream>>>(u, tmp16);
  gemm_bn_kernel<<<512, 256, 0, stream>>>(tmp16, wp16, p_bn, u);
  lif_kernel<float, false><<<CH / 256, 256, 0, stream>>>(u, out);
}
